// _EfficientSelfAttn_25718264168519
// MI455X (gfx1250) — matrix operand order repaired and checked by reading, not compiled, not run on hardware
//
#include <hip/hip_runtime.h>
#include <hip/hip_bf16.h>
#include <stdint.h>

typedef _Float16 half16 __attribute__((ext_vector_type(16)));
typedef float    float8 __attribute__((ext_vector_type(8)));
typedef unsigned int u32x4 __attribute__((ext_vector_type(4)));
typedef int          i32x8 __attribute__((ext_vector_type(8)));
typedef int          i32x4 __attribute__((ext_vector_type(4)));

#define NB   8
#define NQ   16384
#define CD   64
#define NKEY 256

// workspace layout (in _Float16 units)
#define WS_WQ    0          // 64*64        = 4096
#define WS_WKV   4096       // 128*64       = 8192
#define WS_WPROJ 12288      // 64*64       = 4096
#define WS_WSR   16384      // 64*4096     = 262144  (flat [co][ci*64+i*8+j])
#define WS_K     278528     // 8*256*64    = 131072  ([b][m][d] row-major)
#define WS_VT    409600     // 8*64*256    = 131072  ([b][d][m] transposed)

__device__ __forceinline__ float8 wmma_f16(half16 a, half16 b, float8 c) {
  return __builtin_amdgcn_wmma_f32_16x16x32_f16(false, a, false, b, (short)0, c,
                                                false, false);
}

// B-fragment (K x 16, K-chunk of 32): lanes 0-15 -> col n, K=k0..k0+7,k0+16..k0+23;
// lanes 16-31 -> col n, K=k0+8..k0+15,k0+24..k0+31. Storage [n][K] row-major.
__device__ __forceinline__ half16 load_bfrag(const _Float16* base, int row_n,
                                             int stride, int k0, int lane) {
  const _Float16* p = base + row_n * stride + k0 + ((lane & 16) ? 8 : 0);
  half16 r;
#pragma unroll
  for (int h = 0; h < 8; ++h) { r[h] = p[h]; r[h + 8] = p[h + 16]; }
  return r;
}

// A-fragment (16 x 32) from row-major [row][K] storage (ISA 7.12.2 layout).
__device__ __forceinline__ half16 load_afrag(const _Float16* base, int stride,
                                             int k0, int lane) {
  const _Float16* p = base + (lane & 15) * stride + k0 + ((lane & 16) ? 8 : 0);
  half16 r;
#pragma unroll
  for (int h = 0; h < 8; ++h) { r[h] = p[h]; r[h + 8] = p[h + 16]; }
  return r;
}

// ---- 16-lane reductions via DPP16 (VALU, no LDS traffic). The sequence
// pair-swap, quad-rotate, half-mirror, mirror is a valid combining network:
// each step merges groups that are already internally uniform. DPP16 rows of
// 16 lanes keep the wave's two row-groups independent.
template <int CTRL>
__device__ __forceinline__ float dpp_xfer(float v) {
  int i = __builtin_bit_cast(int, v);
  return __builtin_bit_cast(
      float, __builtin_amdgcn_update_dpp(i, i, CTRL, 0xF, 0xF, false));
}
__device__ __forceinline__ float red_max16(float v) {
  v = fmaxf(v, dpp_xfer<0xB1>(v));    // quad_perm [1,0,3,2]
  v = fmaxf(v, dpp_xfer<0x4E>(v));    // quad_perm [2,3,0,1]
  v = fmaxf(v, dpp_xfer<0x141>(v));   // row_half_mirror
  v = fmaxf(v, dpp_xfer<0x140>(v));   // row_mirror
  return v;
}
__device__ __forceinline__ float red_sum16(float v) {
  v += dpp_xfer<0xB1>(v);
  v += dpp_xfer<0x4E>(v);
  v += dpp_xfer<0x141>(v);
  v += dpp_xfer<0x140>(v);
  return v;
}

// ---- Tensor Data Mover: async 2D tile load Global->LDS (ISA ch.8 D# layout).
// group0: count=1 | lds_addr[63:32] | global_addr[120:64] | type=2 [127:126]
// group1: data_size=2B [17:16], tensor_dim0 [79:48], tensor_dim1 [111:80],
//         tile_dim0 [127:112], tile_dim1 [143:128], dim0_stride [207:160]
// This toolchain exposes the 6-arg builtin:
//   (u32x4 g0, i32x8 g1, i32x4 g2, i32x4 g3, i32x8, i32 cpol)
#if __has_builtin(__builtin_amdgcn_tensor_load_to_lds)
#define HAVE_TDM 1
__device__ __forceinline__ void tdm_load_2d(unsigned lds_off, const void* gptr,
                                            unsigned tile_w, unsigned tile_h,
                                            unsigned stride_elems) {
  unsigned long long ga = (unsigned long long)(uintptr_t)gptr;
  u32x4 g0;
  g0[0] = 1u;                                              // count = 1
  g0[1] = lds_off;                                         // LDS byte address
  g0[2] = (unsigned)ga;                                    // gaddr lo32
  g0[3] = ((unsigned)(ga >> 32) & 0x01FFFFFFu) | 0x80000000u;  // gaddr hi|type=2
  i32x8 g1;
  g1[0] = (int)(1u << 16);                                 // data_size = 2 bytes
  g1[1] = (int)((tile_w & 0xFFFFu) << 16);                 // tensor_dim0 lo16
  g1[2] = (int)((tile_w >> 16) | ((tile_h & 0xFFFFu) << 16));  // d0 hi | d1 lo
  g1[3] = (int)((tile_h >> 16) | ((tile_w & 0xFFFFu) << 16));  // d1 hi | tile_dim0
  g1[4] = (int)(tile_h & 0xFFFFu);                         // tile_dim1 (dim2=0)
  g1[5] = (int)stride_elems;                               // dim0_stride lo32
  g1[6] = 0;                                               // stride hi bits
  g1[7] = 0;
  i32x4 z4 = {0, 0, 0, 0};
  i32x8 z8 = {0, 0, 0, 0, 0, 0, 0, 0};
  __builtin_amdgcn_tensor_load_to_lds(g0, g1, z4, z4, z8, 0);
}
#else
#define HAVE_TDM 0
#endif

// ---------------------------------------------------------------------------
// Kernel 1: weights -> f16 workspace (wsr's natural (co,ci,i,j) order IS the
// [co][k] GEMM-B layout with k = ci*64+i*8+j).
// ---------------------------------------------------------------------------
__global__ __launch_bounds__(256) void prep_kernel(const float* wq,
                                                   const float* wkv,
                                                   const float* wproj,
                                                   const float* wsr,
                                                   _Float16* ws) {
  int i = blockIdx.x * blockDim.x + threadIdx.x;
  if (i < 4096)   ws[WS_WQ + i]    = (_Float16)wq[i];
  if (i < 8192)   ws[WS_WKV + i]   = (_Float16)wkv[i];
  if (i < 4096)   ws[WS_WPROJ + i] = (_Float16)wproj[i];
  if (i < 262144) ws[WS_WSR + i]   = (_Float16)wsr[i];
}

// ---------------------------------------------------------------------------
// Kernel 2: KV build. Per wave: one 16-patch row tile.
//   conv-as-GEMM (16x4096 @ 4096x64) -> LayerNorm -> kv proj (16x64 @ 64x128)
// grid = 16 blocks (8 batches x 2 halves), 256 threads (8 waves).
// ---------------------------------------------------------------------------
__global__ __launch_bounds__(256) void kv_kernel(const float* __restrict__ x,
                                                 const float* __restrict__ bkv,
                                                 const float* __restrict__ gamma,
                                                 const float* __restrict__ beta,
                                                 const _Float16* __restrict__ ws,
                                                 _Float16* __restrict__ k_out,
                                                 _Float16* __restrict__ vT_out) {
  __shared__ _Float16 scratch[8][16 * 64];
  const int b = blockIdx.x >> 1;
  const int wave = threadIdx.x >> 5, lane = threadIdx.x & 31;
  const int rowtile = ((blockIdx.x & 1) << 3) + wave;   // 0..15
  const _Float16* wsr_h = ws + WS_WSR;
  const _Float16* wkv_h = ws + WS_WKV;
  const float* xb = x + (size_t)b * NQ * CD;

  const int pidx = rowtile * 16 + (lane & 15);          // patch 0..255
  const int py = pidx >> 4, px = pidx & 15;

  float8 acc[4];
#pragma unroll
  for (int t = 0; t < 4; ++t) acc[t] = float8{0, 0, 0, 0, 0, 0, 0, 0};

  for (int kc = 0; kc < 128; ++kc) {        // K = 4096 in chunks of 32
    const int k0 = kc * 32;
    const int kb = k0 + ((lane & 16) ? 8 : 0);
    half16 a;
#pragma unroll
    for (int h = 0; h < 16; ++h) {
      int k = kb + ((h < 8) ? h : (8 + h));
      int ci = k >> 6, rem = k & 63, ii = rem >> 3, jj = rem & 7;
      a[h] = (_Float16)xb[(size_t)((py * 8 + ii) * 128 + (px * 8 + jj)) * CD + ci];
    }
#pragma unroll
    for (int t = 0; t < 4; ++t) {
      half16 bf = load_bfrag(wsr_h, t * 16 + (lane & 15), 4096, k0, lane);
      acc[t] = wmma_f16(a, bf, acc[t]);
    }
  }

  // LayerNorm over 64 channels (rows r and r+8 live in the two lane halves)
  float mu[8], rstd[8];
#pragma unroll
  for (int r = 0; r < 8; ++r) {
    float s = acc[0][r] + acc[1][r] + acc[2][r] + acc[3][r];
    mu[r] = red_sum16(s) * (1.0f / 64.0f);
    float v = 0.f;
#pragma unroll
    for (int t = 0; t < 4; ++t) { float d = acc[t][r] - mu[r]; v += d * d; }
    v = red_sum16(v) * (1.0f / 64.0f);
    rstd[r] = rsqrtf(v + 1e-5f);
  }

  _Float16* sc = scratch[wave];
#pragma unroll
  for (int t = 0; t < 4; ++t) {
    const int col = t * 16 + (lane & 15);
    const float g = gamma[col], be = beta[col];
#pragma unroll
    for (int r = 0; r < 8; ++r) {
      int row = r + ((lane & 16) ? 8 : 0);
      sc[row * 64 + col] = (_Float16)((acc[t][r] - mu[r]) * rstd[r] * g + be);
    }
  }
  half16 a0 = load_afrag(sc, 64, 0, lane);
  half16 a1 = load_afrag(sc, 64, 32, lane);

  // kv projection: 16x64 @ 64x128 -> k (cols 0..63), v (cols 64..127)
  const int mbase = rowtile * 16;
#pragma unroll
  for (int ct = 0; ct < 8; ++ct) {
    float8 c = float8{0, 0, 0, 0, 0, 0, 0, 0};
    c = wmma_f16(a0, load_bfrag(wkv_h, ct * 16 + (lane & 15), 64, 0, lane), c);
    c = wmma_f16(a1, load_bfrag(wkv_h, ct * 16 + (lane & 15), 64, 32, lane), c);
    const int colg = ct * 16 + (lane & 15);
    const float bb = bkv[colg];
#pragma unroll
    for (int r = 0; r < 8; ++r) {
      int m = mbase + r + ((lane & 16) ? 8 : 0);
      float val = c[r] + bb;
      if (colg < 64)
        k_out[((size_t)b * NKEY + m) * CD + colg] = (_Float16)val;
      else
        vT_out[((size_t)b * CD + (colg - 64)) * NKEY + m] = (_Float16)val;
    }
  }
}

// ---------------------------------------------------------------------------
// Kernel 3: fused q-proj + flash attention + output projection.
// grid = 8 * (16384/128) = 1024 blocks, 256 threads (8 waves x 16 query rows).
// K/V tiles come in through the Tensor Data Mover (wave 0 issues, TENSORcnt
// gates, workgroup barrier releases consumers).
// ---------------------------------------------------------------------------
__global__ __launch_bounds__(256) void attn_kernel(
    const float* __restrict__ x, const float* __restrict__ bq,
    const float* __restrict__ bproj, const _Float16* __restrict__ ws,
    const _Float16* __restrict__ k_all, const _Float16* __restrict__ vT_all,
    float* __restrict__ out) {
  __shared__ _Float16 k_lds[128 * 64];   // 128 keys x 64, row-major
  __shared__ _Float16 vT_lds[64 * 128];  // 64 d x 128 keys
  __shared__ _Float16 scratch[8][16 * 64];

  const int tiles_per_b = NQ / 128;
  const int b = blockIdx.x / tiles_per_b;
  const int qblk = blockIdx.x % tiles_per_b;
  const int wave = threadIdx.x >> 5, lane = threadIdx.x & 31;
  const int qrow = qblk * 128 + wave * 16;

  const _Float16* wq_h = ws + WS_WQ;
  const _Float16* wp_h = ws + WS_WPROJ;
  const _Float16* k_src = k_all + (size_t)b * NKEY * CD;
  const _Float16* v_src = vT_all + (size_t)b * CD * NKEY;

  // ---- q = (x @ wqT + bq) * scale, kept as A-fragments
  half16 qf0, qf1;
  {
    const float* p = x + ((size_t)b * NQ + qrow + (lane & 15)) * CD;
    const int kb = (lane & 16) ? 8 : 0;
    half16 a0, a1;
#pragma unroll
    for (int h = 0; h < 8; ++h) {
      a0[h] = (_Float16)p[kb + h];       a0[h + 8] = (_Float16)p[kb + 16 + h];
      a1[h] = (_Float16)p[32 + kb + h];  a1[h + 8] = (_Float16)p[32 + kb + 16 + h];
    }
    _Float16* sc = scratch[wave];
    const float scale = 0.125f;   // 64^-0.5
#pragma unroll
    for (int t = 0; t < 4; ++t) {
      float8 qc = float8{0, 0, 0, 0, 0, 0, 0, 0};
      qc = wmma_f16(a0, load_bfrag(wq_h, t * 16 + (lane & 15), 64, 0, lane), qc);
      qc = wmma_f16(a1, load_bfrag(wq_h, t * 16 + (lane & 15), 64, 32, lane), qc);
      const int col = t * 16 + (lane & 15);
      const float bqv = bq[col];
#pragma unroll
      for (int r = 0; r < 8; ++r) {
        int row = r + ((lane & 16) ? 8 : 0);
        sc[row * 64 + col] = (_Float16)((qc[r] + bqv) * scale);
      }
    }
    qf0 = load_afrag(sc, 64, 0, lane);
    qf1 = load_afrag(sc, 64, 32, lane);
  }

  // ---- online softmax state
  float mrow[8], lrow[8];
  float8 oacc[4];
#pragma unroll
  for (int r = 0; r < 8; ++r) { mrow[r] = -1e30f; lrow[r] = 0.f; }
#pragma unroll
  for (int t = 0; t < 4; ++t) oacc[t] = float8{0, 0, 0, 0, 0, 0, 0, 0};

  const unsigned k_off = (unsigned)(uintptr_t)(void*)k_lds;
  const unsigned v_off = (unsigned)(uintptr_t)(void*)vT_lds;

  for (int ck = 0; ck < 2; ++ck) {
    __syncthreads();  // previous chunk fully consumed
#if HAVE_TDM
    if (wave == 0) {
      // K chunk: 128 rows x 64 halfs, contiguous (stride 64)
      tdm_load_2d(k_off, k_src + ck * 128 * CD, 64, 128, 64);
      // V^T chunk: 64 rows x 128 halfs out of a 256-half-stride tensor
      tdm_load_2d(v_off, v_src + ck * 128, 128, 64, NKEY);
      __builtin_amdgcn_s_wait_tensorcnt(0);
    }
#else
    {
      const uint4* s = (const uint4*)(k_src + ck * 128 * CD);
      uint4* d = (uint4*)k_lds;
      for (int i = threadIdx.x; i < 1024; i += 256) d[i] = s[i];
      for (int i = threadIdx.x; i < 1024; i += 256) {
        int dd = i >> 4, cc = i & 15;
        ((uint4*)vT_lds)[i] =
            *(const uint4*)(v_src + (size_t)dd * NKEY + ck * 128 + cc * 8);
      }
    }
#endif
    __syncthreads();

#pragma unroll
    for (int sub = 0; sub < 4; ++sub) {
      const int m0 = sub * 32;
      // scores: 16 x 32 (two col tiles), K-dim = 64 in two WMMA steps
      float8 s0 = float8{0, 0, 0, 0, 0, 0, 0, 0};
      float8 s1 = float8{0, 0, 0, 0, 0, 0, 0, 0};
      s0 = wmma_f16(qf0, load_bfrag(k_lds, m0 + (lane & 15), 64, 0, lane), s0);
      s0 = wmma_f16(qf1, load_bfrag(k_lds, m0 + (lane & 15), 64, 32, lane), s0);
      s1 = wmma_f16(qf0, load_bfrag(k_lds, m0 + 16 + (lane & 15), 64, 0, lane), s1);
      s1 = wmma_f16(qf1, load_bfrag(k_lds, m0 + 16 + (lane & 15), 64, 32, lane), s1);

      // online softmax update; write P (16x32 f16) to per-wave scratch
      _Float16* psc = scratch[wave];
      float alpha[8];
#pragma unroll
      for (int r = 0; r < 8; ++r) {
        float cm = red_max16(fmaxf(s0[r], s1[r]));
        float mn = fmaxf(mrow[r], cm);
        alpha[r] = __expf(mrow[r] - mn);
        float p0 = __expf(s0[r] - mn);
        float p1 = __expf(s1[r] - mn);
        lrow[r] = lrow[r] * alpha[r] + red_sum16(p0 + p1);
        mrow[r] = mn;
        int row = r + ((lane & 16) ? 8 : 0);
        psc[row * 32 + (lane & 15)] = (_Float16)p0;
        psc[row * 32 + 16 + (lane & 15)] = (_Float16)p1;
      }
#pragma unroll
      for (int t = 0; t < 4; ++t)
#pragma unroll
        for (int r = 0; r < 8; ++r) oacc[t][r] *= alpha[r];

      half16 pf = load_afrag(psc, 32, 0, lane);
#pragma unroll
      for (int t = 0; t < 4; ++t)
        oacc[t] = wmma_f16(
            pf, load_bfrag(vT_lds, t * 16 + (lane & 15), 128, m0, lane),
            oacc[t]);
    }
  }

  // ---- normalize, output projection, store
  _Float16* sc = scratch[wave];
#pragma unroll
  for (int t = 0; t < 4; ++t) {
    const int col = t * 16 + (lane & 15);
#pragma unroll
    for (int r = 0; r < 8; ++r) {
      int row = r + ((lane & 16) ? 8 : 0);
      sc[row * 64 + col] = (_Float16)(oacc[t][r] / lrow[r]);
    }
  }
  half16 of0 = load_afrag(sc, 64, 0, lane);
  half16 of1 = load_afrag(sc, 64, 32, lane);

  float* orow = out + ((size_t)b * NQ + qrow) * CD;
#pragma unroll
  for (int t = 0; t < 4; ++t) {
    float8 oc = float8{0, 0, 0, 0, 0, 0, 0, 0};
    oc = wmma_f16(of0, load_bfrag(wp_h, t * 16 + (lane & 15), 64, 0, lane), oc);
    oc = wmma_f16(of1, load_bfrag(wp_h, t * 16 + (lane & 15), 64, 32, lane), oc);
    const int col = t * 16 + (lane & 15);
    const float bp = bproj[col];
#pragma unroll
    for (int r = 0; r < 8; ++r) {
      int row = r + ((lane & 16) ? 8 : 0);
      orow[(size_t)row * CD + col] = oc[r] + bp;
    }
  }
}

// ---------------------------------------------------------------------------
extern "C" void kernel_launch(void* const* d_in, const int* in_sizes, int n_in,
                              void* d_out, int out_size, void* d_ws,
                              size_t ws_size, hipStream_t stream) {
  const float* x     = (const float*)d_in[0];
  const float* wq    = (const float*)d_in[1];
  const float* bq    = (const float*)d_in[2];
  const float* wkv   = (const float*)d_in[3];
  const float* bkv   = (const float*)d_in[4];
  const float* wproj = (const float*)d_in[5];
  const float* bproj = (const float*)d_in[6];
  const float* wsr   = (const float*)d_in[7];
  const float* gamma = (const float*)d_in[8];
  const float* beta  = (const float*)d_in[9];
  _Float16* ws = (_Float16*)d_ws;
  _Float16* k_ws = ws + WS_K;
  _Float16* vT_ws = ws + WS_VT;
  float* out = (float*)d_out;

  prep_kernel<<<1024, 256, 0, stream>>>(wq, wkv, wproj, wsr, ws);
  kv_kernel<<<16, 256, 0, stream>>>(x, bkv, gamma, beta, ws, k_ws, vT_ws);
  attn_kernel<<<NB * (NQ / 128), 256, 0, stream>>>(x, bq, bproj, ws, k_ws,
                                                   vT_ws, out);
}
